// LatentModuleLSTM_1116691496966
// MI455X (gfx1250) — compile-verified, matrix operand order checked by reading, not run on hardware
//
#include <hip/hip_runtime.h>

// ---------------------------------------------------------------------------
// Problem constants (match reference setup_inputs)
// ---------------------------------------------------------------------------
#define B_  2048
#define H_  1024
#define D_  1024
#define I_  1024
#define O_  1024
#define L_  24
#define G_  4096   // 4*H

typedef __attribute__((ext_vector_type(16))) __bf16 bf16x16;
typedef __attribute__((ext_vector_type(8)))  __bf16 bf16x8;
typedef __attribute__((ext_vector_type(8)))  float  f32x8;

// ---------------------------------------------------------------------------
// Fragment load: CDNA5 16x32 bf16 A/B operand layout.
// lane<16  : row M=lane,    K = {k..k+7, k+16..k+23}
// lane>=16 : row M=lane-16, K = {k+8..k+15, k+24..k+31}
// Caller bakes the +8 half-wave K offset into the pointer; we load two
// 16-byte chunks 16 bf16 apart in K.
// ---------------------------------------------------------------------------
__device__ __forceinline__ bf16x16 load_frag16(const __bf16* p) {
  bf16x8 lo = *(const bf16x8*)p;
  bf16x8 hi = *(const bf16x8*)(p + 16);
  bf16x16 r;
#pragma unroll
  for (int i = 0; i < 8; ++i) { r[i] = lo[i]; r[i + 8] = hi[i]; }
  return r;
}

#define WMMA_BF16(A, Bm, Cm) \
  __builtin_amdgcn_wmma_f32_16x16x32_bf16(false, (A), false, (Bm), (short)0, (Cm), false, false)

// ---------------------------------------------------------------------------
// C[M,N] (f32, row stride ldc) = A0 @ W0^T (+ A1 @ W1^T) + bias
// A: [M,K] bf16 row-major, W: [N,K] bf16 row-major (i.e. B^T layout).
// Requires M%64==0, N%256==0, K%32==0. Block: 256 thr = 8 waves,
// wave tile 32x64 (2x4 wmma accumulators), block tile 64x256.
// ---------------------------------------------------------------------------
__global__ __launch_bounds__(256)
void wmma_gemm_bf16(const __bf16* __restrict__ A0, const __bf16* __restrict__ W0,
                    const __bf16* __restrict__ A1, const __bf16* __restrict__ W1,
                    const float* __restrict__ bias, float* __restrict__ C,
                    int M, int N, int K, long ldc)
{
  const int lane = threadIdx.x & 31;
  const int wave = threadIdx.x >> 5;
  const int wm   = wave & 1;     // 2 M slots
  const int wn   = wave >> 1;    // 4 N slots
  const int m0   = blockIdx.y * 64  + wm * 32;
  const int n0   = blockIdx.x * 256 + wn * 64;
  if (m0 >= M || n0 >= N) return;

  const int l15  = lane & 15;
  const int koff = (lane >> 4) * 8;

  f32x8 acc[2][4] = {};

  const int npass = (A1 != nullptr) ? 2 : 1;
  for (int pass = 0; pass < npass; ++pass) {
    const __bf16* A = pass ? A1 : A0;
    const __bf16* W = pass ? W1 : W0;
    const __bf16* a0p = A + (long)(m0 +      l15) * K + koff;
    const __bf16* a1p = A + (long)(m0 + 16 + l15) * K + koff;
    const __bf16* w0p = W + (long)(n0 +      l15) * K + koff;
    const __bf16* w1p = W + (long)(n0 + 16 + l15) * K + koff;
    const __bf16* w2p = W + (long)(n0 + 32 + l15) * K + koff;
    const __bf16* w3p = W + (long)(n0 + 48 + l15) * K + koff;
    for (int k = 0; k < K; k += 32) {
      bf16x16 a0 = load_frag16(a0p + k);
      bf16x16 a1 = load_frag16(a1p + k);
      bf16x16 b0 = load_frag16(w0p + k);
      bf16x16 b1 = load_frag16(w1p + k);
      bf16x16 b2 = load_frag16(w2p + k);
      bf16x16 b3 = load_frag16(w3p + k);
      acc[0][0] = WMMA_BF16(a0, b0, acc[0][0]);
      acc[1][0] = WMMA_BF16(a1, b0, acc[1][0]);
      acc[0][1] = WMMA_BF16(a0, b1, acc[0][1]);
      acc[1][1] = WMMA_BF16(a1, b1, acc[1][1]);
      acc[0][2] = WMMA_BF16(a0, b2, acc[0][2]);
      acc[1][2] = WMMA_BF16(a1, b2, acc[1][2]);
      acc[0][3] = WMMA_BF16(a0, b3, acc[0][3]);
      acc[1][3] = WMMA_BF16(a1, b3, acc[1][3]);
    }
  }

  // Epilogue: C/D layout -> lane<16 rows r, lane>=16 rows 8+r, col = lane&15.
  const long rbase = m0 + ((lane >> 4) << 3);
#pragma unroll
  for (int t = 0; t < 4; ++t) {
    const int n = n0 + t * 16 + l15;
    const float bv = bias ? bias[n] : 0.0f;
#pragma unroll
    for (int mi = 0; mi < 2; ++mi) {
      const long row0 = rbase + mi * 16;
#pragma unroll
      for (int r = 0; r < 8; ++r)
        C[(row0 + r) * ldc + n] = acc[mi][t][r] + bv;
    }
  }
}

// ---------------------------------------------------------------------------
// Pointwise / bookkeeping kernels
// ---------------------------------------------------------------------------
__global__ void k_cvt_bf16(const float* __restrict__ s, __bf16* __restrict__ d, int n) {
  int i = blockIdx.x * blockDim.x + threadIdx.x;
  if (i < n) d[i] = (__bf16)s[i];
}
__global__ void k_fill_bf16(__bf16* d, int n) {
  int i = blockIdx.x * blockDim.x + threadIdx.x;
  if (i < n) d[i] = (__bf16)0.0f;
}
__global__ void k_fill_f32(float* d, int n) {
  int i = blockIdx.x * blockDim.x + threadIdx.x;
  if (i < n) d[i] = 0.0f;
}
__global__ void k_fill_i32(int* d, int n) {
  int i = blockIdx.x * blockDim.x + threadIdx.x;
  if (i < n) d[i] = 0;
}
__global__ void k_add_bias(const float* __restrict__ a, const float* __restrict__ b,
                           float* __restrict__ o, int n) {
  int i = blockIdx.x * blockDim.x + threadIdx.x;
  if (i < n) o[i] = a[i] + b[i];
}
__global__ void k_copy_i32(const int* __restrict__ s, int* __restrict__ d, int n) {
  int i = blockIdx.x * blockDim.x + threadIdx.x;
  if (i < n) d[i] = s[i];
}

__device__ __forceinline__ float sigm(float x) { return 1.0f / (1.0f + expf(-x)); }

// gates [B,4H] (i,f,g,o) -> update hx (f32 + bf16 copy) and cx in place.
__global__ void k_lstm_cell(const float* __restrict__ gates,
                            float* __restrict__ hx, float* __restrict__ cx,
                            __bf16* __restrict__ hx_bf) {
  int i = blockIdx.x * blockDim.x + threadIdx.x;
  if (i >= B_ * H_) return;
  int b = i / H_, h = i - b * H_;
  const float* g = gates + (long)b * G_;
  float ig = g[h], fg = g[H_ + h], gg = g[2 * H_ + h], og = g[3 * H_ + h];
  float c  = sigm(fg) * cx[i] + sigm(ig) * tanhf(gg);
  float hv = sigm(og) * tanhf(c);
  cx[i] = c; hx[i] = hv; hx_bf[i] = (__bf16)hv;
}

// One block per batch row: arg = argmax(logits + gumbel) (first-max tie break),
// z row = one-hot (hard / frozen-EOS), update fin & eos_ind, record idx.
__global__ __launch_bounds__(256)
void k_gumbel_argmax(const float* __restrict__ logits_base,  // [B,L,D]
                     const float* __restrict__ gumbel,       // [L,B,D]
                     float* __restrict__ z_base,             // [B,L,D]
                     int* __restrict__ idx_seq,              // [L,B]
                     int* __restrict__ fin, int* __restrict__ eos_ind, int t)
{
  __shared__ float smax[256];
  __shared__ int   sidx[256];
  __shared__ int   eff_s;
  const int b = blockIdx.x;
  const float* lrow = logits_base + ((long)b * L_ + t) * D_;
  const float* grow = gumbel + ((long)t * B_ + b) * D_;
  float best = -3.4e38f; int bidx = 0;
  for (int d = threadIdx.x; d < D_; d += blockDim.x) {
    float v = lrow[d] + grow[d];
    if (v > best) { best = v; bidx = d; }
  }
  smax[threadIdx.x] = best; sidx[threadIdx.x] = bidx;
  __syncthreads();
  for (int s = 128; s > 0; s >>= 1) {
    if (threadIdx.x < s) {
      float v2 = smax[threadIdx.x + s]; int i2 = sidx[threadIdx.x + s];
      if (v2 > smax[threadIdx.x] ||
          (v2 == smax[threadIdx.x] && i2 < sidx[threadIdx.x])) {
        smax[threadIdx.x] = v2; sidx[threadIdx.x] = i2;
      }
    }
    __syncthreads();
  }
  if (threadIdx.x == 0) {
    int arg = sidx[0];
    int fb  = fin[b];
    int eff = fb ? 0 : arg;          // frozen rows emit EOS one-hot
    idx_seq[(long)t * B_ + b] = eff;
    int fa = fb || (arg == 0);
    fin[b] = fa;
    eos_ind[b] += fa ? 0 : 1;
    eff_s = eff;
  }
  __syncthreads();
  float* zrow = z_base + ((long)b * L_ + t) * D_;
  for (int d = threadIdx.x; d < D_; d += blockDim.x)
    zrow[d] = (d == eff_s) ? 1.0f : 0.0f;
}

// z is one-hot => z @ w_t2h^T + b_t2h is a column gather (exact fp32 math).
// idx == nullptr means EOS embedding (index 0).
__global__ void k_embed_gather(const float* __restrict__ w_t2h,  // [H,D]
                               const float* __restrict__ b_t2h,
                               const int* __restrict__ idx,
                               __bf16* __restrict__ out) {       // [B,H] bf16
  int i = blockIdx.x * blockDim.x + threadIdx.x;
  if (i >= B_ * H_) return;
  int b = i / H_, h = i - b * H_;
  int id = idx ? idx[b] : 0;
  out[i] = (__bf16)(w_t2h[(long)h * D_ + id] + b_t2h[h]);
}

// Online selection of outs[:, eos_ind]: after decoder step s copy matching rows.
__global__ void k_select_rows(const float* __restrict__ hx,
                              __bf16* __restrict__ sel_bf,
                              const int* __restrict__ eos_ind, int s) {
  int i = blockIdx.x * blockDim.x + threadIdx.x;
  if (i >= B_ * H_) return;
  int b = i / H_;
  if (eos_ind[b] == s) sel_bf[i] = (__bf16)hx[i];
}

// ---------------------------------------------------------------------------
// Host orchestration
// ---------------------------------------------------------------------------
extern "C" void kernel_launch(void* const* d_in, const int* in_sizes, int n_in,
                              void* d_out, int out_size, void* d_ws, size_t ws_size,
                              hipStream_t stream) {
  (void)in_sizes; (void)n_in; (void)out_size; (void)ws_size;

  const float* x       = (const float*)d_in[0];
  const float* gumbel  = (const float*)d_in[1];
  const float* w_in    = (const float*)d_in[2];
  const float* b_in    = (const float*)d_in[3];
  const float* w_out   = (const float*)d_in[4];
  const float* b_out   = (const float*)d_in[5];
  const float* enc_wih = (const float*)d_in[6];
  const float* enc_whh = (const float*)d_in[7];
  const float* enc_bih = (const float*)d_in[8];
  const float* enc_bhh = (const float*)d_in[9];
  const float* dec_wih = (const float*)d_in[10];
  const float* dec_whh = (const float*)d_in[11];
  const float* dec_bih = (const float*)d_in[12];
  const float* dec_bhh = (const float*)d_in[13];
  const float* w_h2t   = (const float*)d_in[14];
  const float* b_h2t   = (const float*)d_in[15];
  const float* w_t2h   = (const float*)d_in[16];
  const float* b_t2h   = (const float*)d_in[17];

  // d_out layout: output [B,O] | z [B,L,D] | logits [B,L,D] | eos_ind [B] (i32)
  float* out_f       = (float*)d_out;
  float* z_base      = out_f + (size_t)B_ * O_;
  float* logits_base = z_base + (size_t)B_ * L_ * D_;
  int*   eos_out     = (int*)(logits_base + (size_t)B_ * L_ * D_);

  // Workspace carve-up
  char* w = (char*)d_ws;
  size_t off = 0;
  auto alloc = [&](size_t bytes) -> void* {
    void* p = w + off;
    off = (off + bytes + 255) & ~(size_t)255;
    return p;
  };
  __bf16* enc_wih_bf = (__bf16*)alloc((size_t)G_ * H_ * 2);
  __bf16* enc_whh_bf = (__bf16*)alloc((size_t)G_ * H_ * 2);
  __bf16* dec_wih_bf = (__bf16*)alloc((size_t)G_ * H_ * 2);
  __bf16* dec_whh_bf = (__bf16*)alloc((size_t)G_ * H_ * 2);
  __bf16* w_h2t_bf   = (__bf16*)alloc((size_t)D_ * H_ * 2);
  __bf16* w_in_bf    = (__bf16*)alloc((size_t)H_ * I_ * 2);
  __bf16* w_out_bf   = (__bf16*)alloc((size_t)O_ * H_ * 2);
  __bf16* x_bf       = (__bf16*)alloc((size_t)B_ * I_ * 2);
  __bf16* hx_bf      = (__bf16*)alloc((size_t)B_ * H_ * 2);
  __bf16* in_bf      = (__bf16*)alloc((size_t)B_ * H_ * 2);
  __bf16* sel_bf     = (__bf16*)alloc((size_t)B_ * H_ * 2);
  float*  hx_f       = (float*)alloc((size_t)B_ * H_ * 4);
  float*  cx_f       = (float*)alloc((size_t)B_ * H_ * 4);
  float*  gates_f    = (float*)alloc((size_t)B_ * G_ * 4);
  float*  enc_bias   = (float*)alloc((size_t)G_ * 4);
  float*  dec_bias   = (float*)alloc((size_t)G_ * 4);
  int*    fin        = (int*)alloc((size_t)B_ * 4);
  int*    eos_ind    = (int*)alloc((size_t)B_ * 4);
  int*    idx_seq    = (int*)alloc((size_t)L_ * B_ * 4);

  const int TPB = 256;
  auto blk = [&](long n) { return dim3((unsigned)((n + TPB - 1) / TPB)); };
  const long nGH = (long)G_ * H_, nDH = (long)D_ * H_, nBH = (long)B_ * H_;

  // --- one-time conversions / init ---
  k_cvt_bf16<<<blk(nGH), TPB, 0, stream>>>(enc_wih, enc_wih_bf, (int)nGH);
  k_cvt_bf16<<<blk(nGH), TPB, 0, stream>>>(enc_whh, enc_whh_bf, (int)nGH);
  k_cvt_bf16<<<blk(nGH), TPB, 0, stream>>>(dec_wih, dec_wih_bf, (int)nGH);
  k_cvt_bf16<<<blk(nGH), TPB, 0, stream>>>(dec_whh, dec_whh_bf, (int)nGH);
  k_cvt_bf16<<<blk(nDH), TPB, 0, stream>>>(w_h2t, w_h2t_bf, (int)nDH);
  k_cvt_bf16<<<blk(nDH), TPB, 0, stream>>>(w_in,  w_in_bf,  (int)nDH);
  k_cvt_bf16<<<blk(nDH), TPB, 0, stream>>>(w_out, w_out_bf, (int)nDH);
  k_cvt_bf16<<<blk(nBH), TPB, 0, stream>>>(x, x_bf, (int)nBH);
  k_add_bias<<<blk(G_), TPB, 0, stream>>>(enc_bih, enc_bhh, enc_bias, G_);
  k_add_bias<<<blk(G_), TPB, 0, stream>>>(dec_bih, dec_bhh, dec_bias, G_);
  k_fill_bf16<<<blk(nBH), TPB, 0, stream>>>(hx_bf, (int)nBH);
  k_fill_bf16<<<blk(nBH), TPB, 0, stream>>>(in_bf, (int)nBH);
  k_fill_f32 <<<blk(nBH), TPB, 0, stream>>>(hx_f, (int)nBH);
  k_fill_i32 <<<blk(B_), TPB, 0, stream>>>(fin, B_);
  k_fill_i32 <<<blk(B_), TPB, 0, stream>>>(eos_ind, B_);

  // cx0 = x @ w_in^T + b_in
  wmma_gemm_bf16<<<dim3(H_ / 256, B_ / 64), TPB, 0, stream>>>(
      x_bf, w_in_bf, nullptr, nullptr, b_in, cx_f, B_, H_, I_, (long)H_);

  // --- encoder ---
  for (int t = 0; t < L_; ++t) {
    wmma_gemm_bf16<<<dim3(G_ / 256, B_ / 64), TPB, 0, stream>>>(
        in_bf, enc_wih_bf, hx_bf, enc_whh_bf, enc_bias, gates_f, B_, G_, H_, (long)G_);
    k_lstm_cell<<<blk(nBH), TPB, 0, stream>>>(gates_f, hx_f, cx_f, hx_bf);
    wmma_gemm_bf16<<<dim3(D_ / 256, B_ / 64), TPB, 0, stream>>>(
        hx_bf, w_h2t_bf, nullptr, nullptr, b_h2t,
        logits_base + (size_t)t * D_, B_, D_, H_, (long)L_ * D_);
    k_gumbel_argmax<<<dim3(B_), TPB, 0, stream>>>(
        logits_base, gumbel, z_base, idx_seq, fin, eos_ind, t);
    k_embed_gather<<<blk(nBH), TPB, 0, stream>>>(
        w_t2h, b_t2h, idx_seq + (size_t)t * B_, in_bf);
  }

  // --- decoder ---
  k_fill_f32 <<<blk(nBH), TPB, 0, stream>>>(hx_f, (int)nBH);
  k_fill_f32 <<<blk(nBH), TPB, 0, stream>>>(cx_f, (int)nBH);
  k_fill_bf16<<<blk(nBH), TPB, 0, stream>>>(hx_bf, (int)nBH);
  for (int s = 0; s <= L_; ++s) {
    const int* idxp = (s < L_) ? (idx_seq + (size_t)s * B_) : nullptr;  // final step = EOS
    k_embed_gather<<<blk(nBH), TPB, 0, stream>>>(w_t2h, b_t2h, idxp, in_bf);
    wmma_gemm_bf16<<<dim3(G_ / 256, B_ / 64), TPB, 0, stream>>>(
        in_bf, dec_wih_bf, hx_bf, dec_whh_bf, dec_bias, gates_f, B_, G_, H_, (long)G_);
    k_lstm_cell<<<blk(nBH), TPB, 0, stream>>>(gates_f, hx_f, cx_f, hx_bf);
    k_select_rows<<<blk(nBH), TPB, 0, stream>>>(hx_f, sel_bf, eos_ind, s);
  }

  // output = selected @ w_out^T + b_out
  wmma_gemm_bf16<<<dim3(O_ / 256, B_ / 64), TPB, 0, stream>>>(
      sel_bf, w_out_bf, nullptr, nullptr, b_out, out_f, B_, O_, H_, (long)O_);

  k_copy_i32<<<blk(B_), TPB, 0, stream>>>(eos_ind, eos_out, B_);
}
